// AttentionBlock_17557826306054
// MI455X (gfx1250) — matrix operand order repaired and checked by reading, not compiled, not run on hardware
//
#include <hip/hip_runtime.h>
#include <hip/hip_bf16.h>
#include <stdint.h>

// ---------------------------------------------------------------------------
// AttentionBlock for MI455X (gfx1250): f16 WMMA flash-attention pipeline.
//   x:[4,256,64,64] f32, w_qkv:[768,256], w_out:[256,256], b_out:[256]
//   out = x + w_out @ attn(w_qkv @ x) + b_out
// GEMMs use v_wmma_f32_16x16x32_f16 (wave32). Attention is streamed (online
// softmax). K/V tiles are staged into LDS once per workgroup with the CDNA5
// async-DMA path (global_load_async_to_lds_b128 + s_wait_asynccnt), double
// buffered so DMA overlaps the WMMA/softmax work.
// Workspace usage: ~42.5 MB (f16 copies of x/w, Q,K,V^T, attn-out).
// ---------------------------------------------------------------------------

typedef _Float16 f16;
typedef __attribute__((ext_vector_type(16))) _Float16 v16h;
typedef __attribute__((ext_vector_type(8)))  _Float16 v8h;
typedef __attribute__((ext_vector_type(8)))  float    v8f;

constexpr int Bn    = 4;
constexpr int Cc    = 256;
constexpr int HW    = 4096;   // 64*64
constexpr int HEADS = 4;
constexpr int Dd    = 64;

__device__ __forceinline__ v8f wmma_f16(v16h a, v16h b, v8f c) {
  return __builtin_amdgcn_wmma_f32_16x16x32_f16(false, a, false, b, (short)0, c,
                                                false, false);
}

// A fragment 16x32 (MxK), row-major memory, p = base + row0*ld + k0.
// Lanes 0-15 hold M=lane, K 0..15; lanes 16-31 hold M=lane-16, K 16..31.
__device__ __forceinline__ v16h load_a16(const f16* p, int ld) {
  const int l = threadIdx.x & 31, m = l & 15, hi = l >> 4;
  const f16* q = p + m * ld + hi * 16;
  v8h lo = *(const v8h*)q;
  v8h hc = *(const v8h*)(q + 8);
  v16h r;
#pragma unroll
  for (int i = 0; i < 8; ++i) { r[i] = lo[i]; r[i + 8] = hc[i]; }
  return r;
}

// B fragment 32x16 (KxN) where memory holds B^T (N-major, K contiguous):
// p = base + n0*ld + k0. Lane n = l%16, K = (l>>4)*16 .. +15 contiguous.
__device__ __forceinline__ v16h load_bt16(const f16* p, int ld) {
  const int l = threadIdx.x & 31, m = l & 15, hi = l >> 4;
  return *(const v16h*)(p + m * ld + hi * 16);
}

// CDNA5 async DMA: copy 16 bytes global -> LDS, tracked by ASYNCcnt.
__device__ __forceinline__ void async_cp16(uint32_t lds_off, const f16* g) {
  asm volatile("global_load_async_to_lds_b128 %0, %1, off"
               :: "v"(lds_off), "v"((unsigned long long)(uintptr_t)g)
               : "memory");
}

// ---------------------------------------------------------------------------
// Elementwise converts
// ---------------------------------------------------------------------------
__global__ void cvt_f32_f16(const float* __restrict__ in, f16* __restrict__ o, int n) {
  int i = blockIdx.x * blockDim.x + threadIdx.x;
  if (i < n) o[i] = (f16)in[i];
}

// x:[b,c,hw] f32 -> Xh:[b,hw,c] f16   (c-contiguous so GEMM B-frags are contiguous)
__global__ void cvt_x_transpose(const float* __restrict__ x, f16* __restrict__ Xh) {
  int i = blockIdx.x * blockDim.x + threadIdx.x;   // B*HW*C threads
  int c   = i & (Cc - 1);
  int pos = (i >> 8) & (HW - 1);
  int b   = i >> 20;
  Xh[i] = (f16)x[((size_t)b * Cc + c) * HW + pos];
}

// ---------------------------------------------------------------------------
// QKV projection: qkv[o,pos] = sum_c w_qkv[o,c] * x[c,pos]   (per batch)
// Writes Q (scaled by d^-1/2) and K as [b,h,pos,d], V transposed [b,h,d,pos].
// Grid (48 o-tiles, 32 pos-blocks, 4 batches), 256 thr = 8 waves, wave=pos tile
// ---------------------------------------------------------------------------
__global__ void qkv_gemm(const f16* __restrict__ Wq, const f16* __restrict__ Xh,
                         f16* __restrict__ Qh, f16* __restrict__ Kh,
                         f16* __restrict__ VTh) {
  const int ot = blockIdx.x, pb = blockIdx.y, b = blockIdx.z;
  const int wave = threadIdx.x >> 5;
  const int l = threadIdx.x & 31, lm = l & 15, hi = l >> 4;
  const int o0 = ot * 16;
  const int pos0 = pb * 128 + wave * 16;
  const f16* Xb = Xh + (size_t)b * HW * Cc;

  v8f acc = {};
#pragma unroll
  for (int c0 = 0; c0 < Cc; c0 += 32) {
    v16h a  = load_a16(Wq + (size_t)o0 * Cc + c0, Cc);
    v16h bt = load_bt16(Xb + (size_t)pos0 * Cc + c0, Cc);
    acc = wmma_f16(a, bt, acc);
  }

  const int sec   = o0 >> 8;       // 0=q 1=k 2=v
  const int ol    = o0 & 255;
  const int head  = ol >> 6;
  const int dbase = ol & 63;
  const int pos   = pos0 + lm;     // C-layout: col N = lane%16
  const size_t bh = (size_t)(b * HEADS + head);

  if (sec < 2) {
    const float s = (sec == 0) ? 0.125f : 1.0f;   // d^-1/2 folded into Q
    v8h pk;
#pragma unroll
    for (int r = 0; r < 8; ++r) pk[r] = (f16)(acc[r] * s);  // rows d contiguous
    f16* dst = (sec == 0 ? Qh : Kh) + ((bh * HW + pos) * Dd + dbase + hi * 8);
    *(v8h*)dst = pk;
  } else {
#pragma unroll
    for (int r = 0; r < 8; ++r) {
      int d = dbase + hi * 8 + r;
      VTh[(bh * Dd + d) * HW + pos] = (f16)acc[r];
    }
  }
}

// ---------------------------------------------------------------------------
// Flash attention: 8 waves per WG, each owns a 16-row i-tile; the WG streams
// j in steps of 32, staging K (32x64) and V^T (64x32) tiles into LDS via the
// async DMA path, double buffered. All B-fragments then come from LDS.
// Oh written as [b,pos,c] f16 (c-contiguous for the projection GEMM B-frags).
// Grid (32 i-blocks, 4 heads, 4 batches), 256 thr = 8 waves.
// ---------------------------------------------------------------------------
__global__ void flash_attn(const f16* __restrict__ Qh, const f16* __restrict__ Kh,
                           const f16* __restrict__ VTh, f16* __restrict__ Oh) {
  __shared__ __align__(16) f16 sK[2][32 * 64];      // [buf][j][d]   4 KB each
  __shared__ __align__(16) f16 sV[2][64 * 32];      // [buf][d][j]   4 KB each
  __shared__ __align__(16) f16 lds_p[8][16 * 32];   // per-wave P tile (16x32)

  const int ib = blockIdx.x, head = blockIdx.y, b = blockIdx.z;
  const int wave = threadIdx.x >> 5;
  const int l = threadIdx.x & 31, lm = l & 15, hi = l >> 4;
  const size_t bh = (size_t)(b * HEADS + head);
  const f16* Q  = Qh  + bh * HW * Dd;
  const f16* K  = Kh  + bh * HW * Dd;
  const f16* VT = VTh + bh * Dd * HW;
  const int i0 = ib * 128 + wave * 16;

  // LDS byte offsets of the staging buffers (AS(3) pointers are offsets).
  const uint32_t kOff[2] = { (uint32_t)(uintptr_t)&sK[0][0],
                             (uint32_t)(uintptr_t)&sK[1][0] };
  const uint32_t vOff[2] = { (uint32_t)(uintptr_t)&sV[0][0],
                             (uint32_t)(uintptr_t)&sV[1][0] };

  // Each of the 256 threads issues exactly 2 async 16-byte copies per tile.
  const int t = threadIdx.x;
  const int krow = t >> 3, kchk = t & 7;   // K: 32 rows x 8 chunks
  const int vrow = t >> 2, vchk = t & 3;   // V: 64 rows x 4 chunks
  auto stage = [&](int j0, int buf) {
    async_cp16(kOff[buf] + (uint32_t)((krow * 64 + kchk * 8) * 2),
               K + (size_t)(j0 + krow) * Dd + kchk * 8);
    async_cp16(vOff[buf] + (uint32_t)((vrow * 32 + vchk * 8) * 2),
               VT + (size_t)vrow * HW + j0 + vchk * 8);
  };

  const v16h aq0 = load_a16(Q + (size_t)i0 * Dd + 0,  Dd);
  const v16h aq1 = load_a16(Q + (size_t)i0 * Dd + 32, Dd);

  v8f o0a = {}, o1a = {}, o2a = {}, o3a = {};
  float mrow[8], lrow[8];
#pragma unroll
  for (int r = 0; r < 8; ++r) { mrow[r] = -1e30f; lrow[r] = 0.0f; }

  stage(0, 0);                                   // prologue: tile 0

  constexpr int NT = HW / 32;                    // 128 j-tiles
  for (int jt = 0; jt < NT; ++jt) {
    const int buf = jt & 1;
    if (jt + 1 < NT) {
      stage((jt + 1) * 32, buf ^ 1);             // prefetch next tile
      asm volatile("s_wait_asynccnt 0x2" ::: "memory");  // this tile landed
    } else {
      asm volatile("s_wait_asynccnt 0x0" ::: "memory");
    }
    __syncthreads();                             // tile visible to all waves

    // ---- B fragments from LDS ----
    auto kfrag = [&](int sub, int k0) {          // 32x16 frag, cols j-sub-tile
      v16h r;
      const int base = (sub * 16 + lm) * 64 + k0 + hi * 16;
#pragma unroll
      for (int i = 0; i < 16; ++i) r[i] = sK[buf][base + i];
      return r;
    };
    auto vfrag = [&](int dt) {                   // 32x16 frag, cols d-tile
      v16h r;
      const int base = (dt * 16 + lm) * 32 + hi * 16;
#pragma unroll
      for (int i = 0; i < 16; ++i) r[i] = sV[buf][base + i];
      return r;
    };

    // ---- S = Q K^T for two 16-col sub-tiles (K dim = 64 = 2x32) ----
    v8f sa, sb;
    {
      v8f tacc = {};
      tacc = wmma_f16(aq0, kfrag(0, 0),  tacc);
      sa   = wmma_f16(aq1, kfrag(0, 32), tacc);
    }
    {
      v8f tacc = {};
      tacc = wmma_f16(aq0, kfrag(1, 0),  tacc);
      sb   = wmma_f16(aq1, kfrag(1, 32), tacc);
    }

    // ---- online softmax (rows live across 16 lanes of a half-wave) ----
#pragma unroll
    for (int r = 0; r < 8; ++r) {
      float v = fmaxf(sa[r], sb[r]);
#pragma unroll
      for (int mask = 1; mask <= 8; mask <<= 1)
        v = fmaxf(v, __shfl_xor(v, mask, 32));
      float mn   = fmaxf(mrow[r], v);
      float corr = __expf(mrow[r] - mn);
      mrow[r] = mn;
      float pa = __expf(sa[r] - mn);
      float pb = __expf(sb[r] - mn);
      float rs = pa + pb;
#pragma unroll
      for (int mask = 1; mask <= 8; mask <<= 1)
        rs += __shfl_xor(rs, mask, 32);
      lrow[r] = lrow[r] * corr + rs;
      o0a[r] *= corr; o1a[r] *= corr; o2a[r] *= corr; o3a[r] *= corr;

      const int row = r + 8 * hi;                // C-layout row
      lds_p[wave][row * 32 + lm]      = (f16)pa;
      lds_p[wave][row * 32 + 16 + lm] = (f16)pb;
    }

    // cross-lane LDS handoff within the wave: drain DS ops before reload
    asm volatile("s_wait_dscnt 0" ::: "memory");

    // ---- reload P in A-fragment layout (16x32) ----
    v16h ap;
    {
      const int base = lm * 32 + hi * 16;
#pragma unroll
      for (int i = 0; i < 8; ++i) {
        ap[i]     = lds_p[wave][base + i];
        ap[i + 8] = lds_p[wave][base + 8 + i];
      }
    }

    // ---- O += P V  (4 d-tiles of 16 cols) ----
    o0a = wmma_f16(ap, vfrag(0), o0a);
    o1a = wmma_f16(ap, vfrag(1), o1a);
    o2a = wmma_f16(ap, vfrag(2), o2a);
    o3a = wmma_f16(ap, vfrag(3), o3a);

    __syncthreads();                             // done with this buffer
  }

  // ---- normalize and store to Oh[b,pos,c] ----
  float inv[8];
#pragma unroll
  for (int r = 0; r < 8; ++r) inv[r] = 1.0f / lrow[r];
  const int cbase = head * 64 + lm;
#pragma unroll
  for (int r = 0; r < 8; ++r) {
    const int pos = i0 + r + 8 * hi;
    f16* dst = Oh + ((size_t)b * HW + pos) * Cc + cbase;
    dst[0]  = (f16)(o0a[r] * inv[r]);
    dst[16] = (f16)(o1a[r] * inv[r]);
    dst[32] = (f16)(o2a[r] * inv[r]);
    dst[48] = (f16)(o3a[r] * inv[r]);
  }
}

// ---------------------------------------------------------------------------
// Output projection + bias + residual (fp32 out)
// Grid (16 o-tiles, 32 pos-blocks, 4 batches), 8 waves, wave = pos tile.
// ---------------------------------------------------------------------------
__global__ void proj_out(const f16* __restrict__ Wo, const f16* __restrict__ Oh,
                         const float* __restrict__ bout,
                         const float* __restrict__ x, float* __restrict__ out) {
  const int ot = blockIdx.x, pb = blockIdx.y, b = blockIdx.z;
  const int wave = threadIdx.x >> 5;
  const int l = threadIdx.x & 31, lm = l & 15, hi = l >> 4;
  const int o0 = ot * 16;
  const int pos0 = pb * 128 + wave * 16;
  const f16* Ob = Oh + (size_t)b * HW * Cc;

  v8f acc = {};
#pragma unroll
  for (int c0 = 0; c0 < Cc; c0 += 32) {
    v16h a  = load_a16(Wo + (size_t)o0 * Cc + c0, Cc);
    v16h bt = load_bt16(Ob + (size_t)pos0 * Cc + c0, Cc);
    acc = wmma_f16(a, bt, acc);
  }

  const int pos = pos0 + lm;
#pragma unroll
  for (int r = 0; r < 8; ++r) {
    const int o = o0 + r + 8 * hi;
    const size_t idx = ((size_t)b * Cc + o) * HW + pos;
    out[idx] = acc[r] + bout[o] + x[idx];
  }
}

// ---------------------------------------------------------------------------
extern "C" void kernel_launch(void* const* d_in, const int* in_sizes, int n_in,
                              void* d_out, int out_size, void* d_ws, size_t ws_size,
                              hipStream_t stream) {
  (void)in_sizes; (void)n_in; (void)out_size; (void)ws_size;
  const float* x      = (const float*)d_in[0];
  const float* w_qkv  = (const float*)d_in[1];
  const float* w_out  = (const float*)d_in[2];
  const float* b_out  = (const float*)d_in[3];
  float*       out    = (float*)d_out;

  char* ws = (char*)d_ws;
  f16* Xh  = (f16*)ws; ws += (size_t)Bn * HW * Cc * sizeof(f16);        // 8 MB
  f16* Wqh = (f16*)ws; ws += (size_t)3 * Cc * Cc * sizeof(f16);         // 384 KB
  f16* Woh = (f16*)ws; ws += (size_t)Cc * Cc * sizeof(f16);             // 128 KB
  f16* Qh  = (f16*)ws; ws += (size_t)Bn * HEADS * HW * Dd * sizeof(f16);// 8 MB
  f16* Kh  = (f16*)ws; ws += (size_t)Bn * HEADS * HW * Dd * sizeof(f16);// 8 MB
  f16* VTh = (f16*)ws; ws += (size_t)Bn * HEADS * Dd * HW * sizeof(f16);// 8 MB
  f16* Oh  = (f16*)ws; ws += (size_t)Bn * HW * Cc * sizeof(f16);        // 8 MB

  cvt_x_transpose<<<(Bn * HW * Cc) / 256, 256, 0, stream>>>(x, Xh);
  cvt_f32_f16<<<(3 * Cc * Cc) / 256, 256, 0, stream>>>(w_qkv, Wqh, 3 * Cc * Cc);
  cvt_f32_f16<<<(Cc * Cc) / 256, 256, 0, stream>>>(w_out, Woh, Cc * Cc);

  qkv_gemm<<<dim3(48, 32, Bn), 256, 0, stream>>>(Wqh, Xh, Qh, Kh, VTh);
  flash_attn<<<dim3(32, HEADS, Bn), 256, 0, stream>>>(Qh, Kh, VTh, Oh);
  proj_out<<<dim3(16, 32, Bn), 256, 0, stream>>>(Woh, Oh, b_out, x, out);
}
